// scoring_model_33543694582407
// MI455X (gfx1250) — compile-verified, matrix operand order checked by reading, not run on hardware
//
#include <hip/hip_runtime.h>
#include <cstdint>
#include <cstddef>

// ============================================================================
// GVP-GNN forward for MI455X (gfx1250), bf16 WMMA for every GEMM.
//   N=20000 nodes, E=400000 edges, NS=128, NV=2, ES=8, EV=1, IN_S=1433
// K-paddings (multiples of 32 for v_wmma_f32_16x16x32_bf16):
//   lin: 1433->1440 (45)  m0: 269->288 (9)  m1/m2: 130->160 (5)
//   ff0: 132->160 (5)     ff1: 516->544 (17)
// M-blocking: msg kernel 4x16 edges/block, ff kernels 2x16 nodes/block, so
// each packed-B fragment load feeds 4 (resp 2) WMMAs.
// Edge gathers use global_load_async_to_lds_b128 (ASYNCcnt) straight to LDS.
//
// Input flattening (top-level insertion order; params pytree, sorted keys):
//   0 node_s, 1 node_v, 2 edge_index, 3 edge_s, 4 edge_v,
//   5..28 l1: [ff0.(bs,wh,ws,wv), ff1.(bs,wh,ws,wv), ln0_b, ln0_g, ln1_b,
//              ln1_g, m0.(bs,wh,ws,wv), m1.(...), m2.(...)]
//   29..52 l2, 53 lin_b, 54 lin_w, 55 ro_b, 56 ro_w
// ============================================================================

#define NN   20000
#define NE   400000
#define INS  1433

#define KPL  1440
#define KP0  288
#define KP12 160
#define KPF0 160
#define KPF1 544

typedef __attribute__((ext_vector_type(16))) __bf16       v16bf;
typedef __attribute__((ext_vector_type(8)))  float        v8f;
typedef __attribute__((ext_vector_type(4)))  unsigned int u32x4;

union Frag { u32x4 q[2]; v16bf bf; };

__device__ __forceinline__ unsigned short f2bf(float f) {
  unsigned int u = __float_as_uint(f);
  u += 0x7FFFu + ((u >> 16) & 1u);          // round-to-nearest-even
  return (unsigned short)(u >> 16);
}
__device__ __forceinline__ float sigmoidf_(float x) { return 1.0f / (1.0f + __expf(-x)); }
__device__ __forceinline__ float wred32(float x) {
#pragma unroll
  for (int m = 16; m > 0; m >>= 1) x += __shfl_xor(x, m, 32);
  return x;
}
__device__ __forceinline__ v8f zero8() {
  v8f z = {0.f,0.f,0.f,0.f,0.f,0.f,0.f,0.f}; return z;
}

// Async memory->LDS 16B copy per lane (VGLOBAL GLOBAL_LOAD_ASYNC_TO_LDS_B128,
// tracked by ASYNCcnt). Low 32 bits of a generic LDS pointer are the LDS
// byte address (flat aperture truncation).
__device__ __forceinline__ void async_ld_b128(void* lds, const void* g) {
  unsigned loff = (unsigned)(uintptr_t)lds;
  asm volatile("global_load_async_to_lds_b128 %0, %1, off"
               :: "v"(loff), "v"(g) : "memory");
}
__device__ __forceinline__ void wait_async0() {
  asm volatile("s_wait_asynccnt 0" ::: "memory");
}

// ---------------------------------------------------------------------------
// MT stacked 16x16 output tiles sharing one B stream.
// A tiles: bf16 [MT*16 x pitch] in LDS (rows contiguous). B fragment-packed
// bf16 in global (tile order [nt][kt], 512 bf16/tile, per-lane contiguous).
// A fragment per ISA 7.12.2: lane row=lane&15; half=lane>>4 selects K chunks
// [8h,8h+8) and [16+8h,16+8h+8) of the k-tile.
// ---------------------------------------------------------------------------
template <int MT>
__device__ __forceinline__ void gemm_mt(const unsigned short* ldsA, int pitch,
                                        const unsigned short* __restrict__ Bp,
                                        int nt, int nkt, v8f* acc) {
  int lane = threadIdx.x & 31;
  int row = lane & 15, half = lane >> 4;
  const unsigned short* ar0 = ldsA + row * pitch + 8 * half;
  const unsigned short* bp = Bp + ((size_t)nt * nkt * 32 + lane) * 16;
  for (int kt = 0; kt < nkt; ++kt) {
    Frag b;
    b.q[0] = *(const u32x4*)(bp);
    b.q[1] = *(const u32x4*)(bp + 8);
    __builtin_prefetch(bp + 512, 0, 1);     // global_prefetch_b8 next B k-tile
    bp += 512;
#pragma unroll
    for (int mi = 0; mi < MT; ++mi) {
      Frag a;
      const unsigned short* ar = ar0 + mi * 16 * pitch + kt * 32;
      a.q[0] = *(const u32x4*)(ar);
      a.q[1] = *(const u32x4*)(ar + 16);
      acc[mi] = __builtin_amdgcn_wmma_f32_16x16x32_bf16(false, a.bf, false, b.bf,
                                                        (short)0, acc[mi], false, false);
    }
  }
}

// C/D layout per ISA: lane covers n=lane&15; vgpr g <-> m = g + 8*(lane>>4)
__device__ __forceinline__ void acc_to_lds(unsigned short* dst, int pitch, int nbase,
                                           v8f acc, const float* __restrict__ bias,
                                           bool relu) {
  int lane = threadIdx.x & 31;
  int n = lane & 15, hb = lane >> 4;
  float bv = bias[nbase + n];
#pragma unroll
  for (int g = 0; g < 8; ++g) {
    int m = g + 8 * hb;
    float y = acc[g] + bv;
    if (relu) y = fmaxf(y, 0.f);
    dst[m * pitch + nbase + n] = f2bf(y);
  }
}

// ---------------------------------------------------------------------------
// Weight packing: row-major f32 [K x N] -> fragment-native bf16 with Kpad.
// ---------------------------------------------------------------------------
__global__ void k_pack_b(const float* __restrict__ src, unsigned short* __restrict__ dst,
                         int K, int N, int Kpad) {
  int total = Kpad * N;
  int nkt = Kpad >> 5;
  for (int t = blockIdx.x * blockDim.x + threadIdx.x; t < total;
       t += gridDim.x * blockDim.x) {
    int tile = t >> 9, w = t & 511;
    int lane = w >> 4, e = w & 15;
    int i = e >> 1, j = e & 1;
    int nt = tile / nkt, kt = tile - nt * nkt;
    int half = lane >> 4, n = (nt << 4) + (lane & 15);
    int k = (kt << 5) + ((i < 4) ? (8 * half + 2 * i + j)
                                 : (16 + 8 * half + 2 * (i - 4) + j));
    float v = (k < K) ? src[(size_t)k * N + n] : 0.0f;
    dst[t] = f2bf(v);
  }
}

__global__ void k_fill0(float* __restrict__ p, int n) {
  for (int i = blockIdx.x * blockDim.x + threadIdx.x; i < n;
       i += gridDim.x * blockDim.x) p[i] = 0.0f;
}

__global__ void k_deg(const int* __restrict__ eidx, float* __restrict__ cnt) {
  int e = blockIdx.x * blockDim.x + threadIdx.x;
  if (e < NE)
    __hip_atomic_fetch_add(&cnt[eidx[NE + e]], 1.0f,
                           __ATOMIC_RELAXED, __HIP_MEMORY_SCOPE_AGENT);
}

// ---------------------------------------------------------------------------
// Input linear: s = node_s @ lin_w + lin_b     [20000x1433]x[1433x128]
// Coalesced A build: consecutive threads read consecutive f32.
// ---------------------------------------------------------------------------
__global__ void __launch_bounds__(256)
k_lin(const float* __restrict__ X, const unsigned short* __restrict__ Bp,
      const float* __restrict__ bias, float* __restrict__ s_out,
      unsigned short* __restrict__ sbf_out) {
  __shared__ __align__(16) unsigned short A[16 * KPL];
  int i0 = blockIdx.x * 16;
  int t = threadIdx.x;
  for (int f = t; f < 16 * INS; f += 256) {
    int r = f / INS, col = f - r * INS;
    A[r * KPL + col] = f2bf(X[(size_t)i0 * INS + f]);
  }
  for (int f = t; f < 16 * (KPL - INS); f += 256) {
    int r = f / (KPL - INS), c = f % (KPL - INS);
    A[r * KPL + INS + c] = 0;
  }
  __syncthreads();
  int w = t >> 5, lane = t & 31;
  v8f acc[1]; acc[0] = zero8();
  gemm_mt<1>(A, KPL, Bp, w, KPL / 32, acc);
  int n = lane & 15, hb = lane >> 4;
  float bv = bias[w * 16 + n];
#pragma unroll
  for (int g = 0; g < 8; ++g) {
    int m = g + 8 * hb;
    float y = acc[0][g] + bv;
    size_t o = (size_t)(i0 + m) * 128 + w * 16 + n;
    s_out[o] = y;
    sbf_out[o] = f2bf(y);
  }
}

// ---------------------------------------------------------------------------
// Fused edge-message kernel: 64 edges/block (4 M-tiles), 8 waves.
// gather(async->LDS) -> m0 -> m1 -> m2 -> scatter-add. Lanes t<64 run the
// tiny vector-channel GVP chain for edge t in f32 VALU between stages.
// A2 aliases A0 (A0 dead after the m0 GEMM) to stay under 64KB LDS.
// ---------------------------------------------------------------------------
#define MTM 4
struct MsgArgs {
  const unsigned short* sbf; const float* v;
  const int* eidx; const float* es; const float* ev;
  const float *wh0, *wv0, *bs0;
  const float *wh1, *wv1, *bs1;
  const float *wh2, *wv2, *bs2;
  const unsigned short *pw0, *pw1, *pw2;
  float *ds, *dv;
};

__global__ void __launch_bounds__(256) k_msg(MsgArgs P) {
  __shared__ __align__(16) unsigned short A0[MTM * 16 * KP0];   // 36,864 B
  __shared__ __align__(16) unsigned short A1[MTM * 16 * KP12];  // 20,480 B
  __shared__ int dsts[MTM * 16];
  unsigned short* A2 = A0;                 // reuse: A0 dead after m0 GEMM
  int e0 = blockIdx.x * (MTM * 16);
  int t = threadIdx.x, w = t >> 5, lane = t & 31;

  // zero K-pad columns of A0/A1
  for (int j = t; j < MTM * 16 * 19; j += 256) { int r = j / 19, c = j % 19; A0[r * KP0 + 269 + c] = 0; }
  for (int j = t; j < MTM * 16 * 30; j += 256) { int r = j / 30, c = j % 30; A1[r * KP12 + 130 + c] = 0; }

  // gather A0 scalar part [s_src | es | s_dst] -- async copies to LDS
  for (int slot = t; slot < MTM * 16 * 16; slot += 256) {
    int r = slot >> 4, c16 = slot & 15;
    int e = e0 + r;
    int sn = P.eidx[e], dn = P.eidx[NE + e];
    async_ld_b128(A0 + r * KP0 + c16 * 8,       P.sbf + (size_t)sn * 128 + c16 * 8);
    async_ld_b128(A0 + r * KP0 + 136 + c16 * 8, P.sbf + (size_t)dn * 128 + c16 * 8);
    if (c16 == 0)
      for (int j = 0; j < 8; ++j) A0[r * KP0 + 128 + j] = f2bf(P.es[(size_t)e * 8 + j]);
  }
  // vector stage 0: vh0 = mv @ wh0 (5x5); vn0 -> A0 cols 264..268
  float vh0[5][3], vh1[2][3], vh2[2][3], v2f[2][3];
  int mydst = 0;
  if (t < MTM * 16) {
    int e = e0 + t;
    int sn = P.eidx[e]; mydst = P.eidx[NE + e];
    dsts[t] = mydst;
    float mv[5][3];
#pragma unroll
    for (int c = 0; c < 3; ++c) {
      mv[0][c] = P.v[(size_t)sn * 6 + c];
      mv[1][c] = P.v[(size_t)sn * 6 + 3 + c];
      mv[2][c] = P.ev[(size_t)e * 3 + c];
      mv[3][c] = P.v[(size_t)mydst * 6 + c];
      mv[4][c] = P.v[(size_t)mydst * 6 + 3 + c];
    }
#pragma unroll
    for (int h = 0; h < 5; ++h) {
      float ss = 0.f;
#pragma unroll
      for (int c = 0; c < 3; ++c) {
        float a = 0.f;
#pragma unroll
        for (int vv = 0; vv < 5; ++vv) a += mv[vv][c] * P.wh0[vv * 5 + h];
        vh0[h][c] = a; ss += a * a;
      }
      A0[t * KP0 + 264 + h] = f2bf(sqrtf(fmaxf(ss, 1e-8f)));
    }
  }
  wait_async0();            // per-wave: async LDS writes landed
  __syncthreads();

  // ---- m0 GEMM (K=288): relu+bias -> A1 cols 0..127 ----
  v8f acc[MTM];
#pragma unroll
  for (int mi = 0; mi < MTM; ++mi) acc[mi] = zero8();
  gemm_mt<MTM>(A0, KP0, P.pw0, w, KP0 / 32, acc);
#pragma unroll
  for (int mi = 0; mi < MTM; ++mi)
    acc_to_lds(A1 + mi * 16 * KP12, KP12, w * 16, acc[mi], P.bs0, true);
  if (t < MTM * 16) {       // gated v0; vh1 = v0 @ wh1; vn1 -> A1 cols 128..129
    float v0[2][3];
#pragma unroll
    for (int o = 0; o < 2; ++o) {
      float ss = 0.f;
#pragma unroll
      for (int c = 0; c < 3; ++c) {
        float a = 0.f;
#pragma unroll
        for (int h = 0; h < 5; ++h) a += vh0[h][c] * P.wv0[h * 2 + o];
        v0[o][c] = a; ss += a * a;
      }
      float g = sigmoidf_(sqrtf(fmaxf(ss, 1e-8f)));
#pragma unroll
      for (int c = 0; c < 3; ++c) v0[o][c] *= g;
    }
#pragma unroll
    for (int h = 0; h < 2; ++h) {
      float ss = 0.f;
#pragma unroll
      for (int c = 0; c < 3; ++c) {
        float a = v0[0][c] * P.wh1[h] + v0[1][c] * P.wh1[2 + h];
        vh1[h][c] = a; ss += a * a;
      }
      A1[t * KP12 + 128 + h] = f2bf(sqrtf(fmaxf(ss, 1e-8f)));
    }
  }
  __syncthreads();

  // ---- m1 GEMM (K=160): relu+bias -> A2 (aliases A0) ----
#pragma unroll
  for (int mi = 0; mi < MTM; ++mi) acc[mi] = zero8();
  gemm_mt<MTM>(A1, KP12, P.pw1, w, KP12 / 32, acc);
#pragma unroll
  for (int mi = 0; mi < MTM; ++mi)
    acc_to_lds(A2 + mi * 16 * KP12, KP12, w * 16, acc[mi], P.bs1, true);
  for (int j = t; j < MTM * 16 * 30; j += 256) {  // zero A2 pads now (A0 dead)
    int r = j / 30, c = j % 30;
    A2[r * KP12 + 130 + c] = 0;
  }
  if (t < MTM * 16) {       // gated v1; vh2 = v1 @ wh2; vn2 -> A2; v2f (no gate)
    float v1[2][3];
#pragma unroll
    for (int o = 0; o < 2; ++o) {
      float ss = 0.f;
#pragma unroll
      for (int c = 0; c < 3; ++c) {
        float a = vh1[0][c] * P.wv1[o] + vh1[1][c] * P.wv1[2 + o];
        v1[o][c] = a; ss += a * a;
      }
      float g = sigmoidf_(sqrtf(fmaxf(ss, 1e-8f)));
#pragma unroll
      for (int c = 0; c < 3; ++c) v1[o][c] *= g;
    }
#pragma unroll
    for (int h = 0; h < 2; ++h) {
      float ss = 0.f;
#pragma unroll
      for (int c = 0; c < 3; ++c) {
        float a = v1[0][c] * P.wh2[h] + v1[1][c] * P.wh2[2 + h];
        vh2[h][c] = a; ss += a * a;
      }
      A2[t * KP12 + 128 + h] = f2bf(sqrtf(fmaxf(ss, 1e-8f)));
    }
#pragma unroll
    for (int o = 0; o < 2; ++o)
#pragma unroll
      for (int c = 0; c < 3; ++c)
        v2f[o][c] = vh2[0][c] * P.wv2[o] + vh2[1][c] * P.wv2[2 + o];
  }
  __syncthreads();

  // ---- m2 GEMM (K=160): bias, no act; scatter-add (mean divide deferred) ----
#pragma unroll
  for (int mi = 0; mi < MTM; ++mi) acc[mi] = zero8();
  gemm_mt<MTM>(A2, KP12, P.pw2, w, KP12 / 32, acc);
  {
    int n = lane & 15, hb = lane >> 4;
    float bv = P.bs2[w * 16 + n];
#pragma unroll
    for (int mi = 0; mi < MTM; ++mi)
#pragma unroll
      for (int g = 0; g < 8; ++g) {
        int m = g + 8 * hb;
        int did = dsts[mi * 16 + m];
        __hip_atomic_fetch_add(&P.ds[(size_t)did * 128 + w * 16 + n],
                               acc[mi][g] + bv,
                               __ATOMIC_RELAXED, __HIP_MEMORY_SCOPE_AGENT);
      }
  }
  if (t < MTM * 16) {
#pragma unroll
    for (int o = 0; o < 2; ++o)
#pragma unroll
      for (int c = 0; c < 3; ++c)
        __hip_atomic_fetch_add(&P.dv[(size_t)mydst * 6 + o * 3 + c], v2f[o][c],
                               __ATOMIC_RELAXED, __HIP_MEMORY_SCOPE_AGENT);
  }
}

// ---------------------------------------------------------------------------
// LayerNorm + residual (+ optional mean-divide by cnt) + vector renorm.
// One wave per node; in-place safe.
// ---------------------------------------------------------------------------
__global__ void __launch_bounds__(256)
k_ln(const float* __restrict__ s_in, const float* __restrict__ sd,
     const float* __restrict__ v_in, const float* __restrict__ vd,
     const float* __restrict__ cnt, const float* __restrict__ g,
     const float* __restrict__ b, float* __restrict__ s_out,
     unsigned short* __restrict__ sbf_out, float* __restrict__ v_out) {
  int node = (blockIdx.x * blockDim.x + threadIdx.x) >> 5;
  int lane = threadIdx.x & 31;
  if (node >= NN) return;
  float inv = cnt ? 1.0f / fmaxf(cnt[node], 1.0f) : 1.0f;
  float x[4];
#pragma unroll
  for (int k = 0; k < 4; ++k) {
    int c = lane + 32 * k;
    x[k] = s_in[(size_t)node * 128 + c] + sd[(size_t)node * 128 + c] * inv;
  }
  float mu = wred32(x[0] + x[1] + x[2] + x[3]) * (1.0f / 128.0f);
  float vsum = 0.f;
#pragma unroll
  for (int k = 0; k < 4; ++k) { float d = x[k] - mu; vsum += d * d; }
  float var = wred32(vsum) * (1.0f / 128.0f);
  float is = rsqrtf(var + 1e-5f);
#pragma unroll
  for (int k = 0; k < 4; ++k) {
    int c = lane + 32 * k;
    float y = (x[k] - mu) * is * g[c] + b[c];
    s_out[(size_t)node * 128 + c] = y;
    sbf_out[(size_t)node * 128 + c] = f2bf(y);
  }
  if (lane == 0) {
    float vv[6];
#pragma unroll
    for (int j = 0; j < 6; ++j)
      vv[j] = v_in[(size_t)node * 6 + j] + vd[(size_t)node * 6 + j] * inv;
    float n0 = fmaxf(vv[0] * vv[0] + vv[1] * vv[1] + vv[2] * vv[2], 1e-8f);
    float n1 = fmaxf(vv[3] * vv[3] + vv[4] * vv[4] + vv[5] * vv[5], 1e-8f);
    float dn = rsqrtf((n0 + n1) * 0.5f);
#pragma unroll
    for (int j = 0; j < 6; ++j) v_out[(size_t)node * 6 + j] = vv[j] * dn;
  }
}

// ---------------------------------------------------------------------------
// ff0: (128,2) -> (512,4), relu + gated vector out. 32 nodes/block (MT=2).
// ---------------------------------------------------------------------------
struct Ff0Args {
  const unsigned short* sbf; const float* v;
  const float *wh, *wv, *bs;
  const unsigned short* pw;
  unsigned short* hidbf; float* hv;
};
__global__ void __launch_bounds__(256) k_ff0(Ff0Args P) {
  __shared__ __align__(16) unsigned short A[2 * 16 * KPF0];
  int i0 = blockIdx.x * 32, t = threadIdx.x, w = t >> 5, lane = t & 31;
  for (int j = t; j < 32 * 28; j += 256) { int r = j / 28, c = j % 28; A[r * KPF0 + 132 + c] = 0; }
  for (int slot = t; slot < 32 * 16; slot += 256) {
    int r = slot >> 4, c16 = slot & 15;
    *(u32x4*)(A + r * KPF0 + c16 * 8) =
        *(const u32x4*)(P.sbf + (size_t)(i0 + r) * 128 + c16 * 8);
  }
  if (t < 32) {
    int i = i0 + t;
    float vv[2][3], vh[4][3];
#pragma unroll
    for (int c = 0; c < 3; ++c) { vv[0][c] = P.v[(size_t)i * 6 + c]; vv[1][c] = P.v[(size_t)i * 6 + 3 + c]; }
#pragma unroll
    for (int h = 0; h < 4; ++h) {
      float ss = 0.f;
#pragma unroll
      for (int c = 0; c < 3; ++c) {
        float a = vv[0][c] * P.wh[h] + vv[1][c] * P.wh[4 + h];
        vh[h][c] = a; ss += a * a;
      }
      A[t * KPF0 + 128 + h] = f2bf(sqrtf(fmaxf(ss, 1e-8f)));
    }
#pragma unroll
    for (int o = 0; o < 4; ++o) {   // gated vector output -> hv
      float vo[3]; float ss = 0.f;
#pragma unroll
      for (int c = 0; c < 3; ++c) {
        float a = 0.f;
#pragma unroll
        for (int h = 0; h < 4; ++h) a += vh[h][c] * P.wv[h * 4 + o];
        vo[c] = a; ss += a * a;
      }
      float gt = sigmoidf_(sqrtf(fmaxf(ss, 1e-8f)));
#pragma unroll
      for (int c = 0; c < 3; ++c) P.hv[(size_t)i * 12 + o * 3 + c] = vo[c] * gt;
    }
  }
  __syncthreads();
  for (int j = 0; j < 4; ++j) {     // 32 N-tiles of 512 cols, 4 per wave
    int nt = w + 8 * j;
    v8f acc[2]; acc[0] = zero8(); acc[1] = zero8();
    gemm_mt<2>(A, KPF0, P.pw, nt, KPF0 / 32, acc);
    int n = lane & 15, hb = lane >> 4;
    float bv = P.bs[nt * 16 + n];
#pragma unroll
    for (int mi = 0; mi < 2; ++mi)
#pragma unroll
      for (int g = 0; g < 8; ++g) {
        int m = g + 8 * hb;
        float y = fmaxf(acc[mi][g] + bv, 0.f);
        P.hidbf[(size_t)(i0 + mi * 16 + m) * 512 + nt * 16 + n] = f2bf(y);
      }
  }
}

// ---------------------------------------------------------------------------
// ff1: (512,4) -> (128,2), no activations. 32 nodes/block (MT=2).
// ---------------------------------------------------------------------------
struct Ff1Args {
  const unsigned short* hidbf; const float* hv;
  const float *wh, *wv, *bs;
  const unsigned short* pw;
  float* hs; float* hv2;
};
__global__ void __launch_bounds__(256) k_ff1(Ff1Args P) {
  __shared__ __align__(16) unsigned short A[2 * 16 * KPF1];   // 34,816 B
  int i0 = blockIdx.x * 32, t = threadIdx.x, w = t >> 5, lane = t & 31;
  for (int j = t; j < 32 * 28; j += 256) { int r = j / 28, c = j % 28; A[r * KPF1 + 516 + c] = 0; }
  for (int chunk = t; chunk < 32 * 64; chunk += 256) {   // 32 rows x 64 x 16B
    int r = chunk >> 6, c = chunk & 63;
    *(u32x4*)(A + r * KPF1 + c * 8) =
        *(const u32x4*)(P.hidbf + (size_t)(i0 + r) * 512 + c * 8);
  }
  if (t < 32) {
    int i = i0 + t;
    float hvv[4][3], vh[4][3];
#pragma unroll
    for (int v = 0; v < 4; ++v)
#pragma unroll
      for (int c = 0; c < 3; ++c) hvv[v][c] = P.hv[(size_t)i * 12 + v * 3 + c];
#pragma unroll
    for (int h = 0; h < 4; ++h) {
      float ss = 0.f;
#pragma unroll
      for (int c = 0; c < 3; ++c) {
        float a = 0.f;
#pragma unroll
        for (int v = 0; v < 4; ++v) a += hvv[v][c] * P.wh[v * 4 + h];
        vh[h][c] = a; ss += a * a;
      }
      A[t * KPF1 + 512 + h] = f2bf(sqrtf(fmaxf(ss, 1e-8f)));
    }
#pragma unroll
    for (int o = 0; o < 2; ++o)
#pragma unroll
      for (int c = 0; c < 3; ++c) {
        float a = 0.f;
#pragma unroll
        for (int h = 0; h < 4; ++h) a += vh[h][c] * P.wv[h * 2 + o];
        P.hv2[(size_t)i * 6 + o * 3 + c] = a;
      }
  }
  __syncthreads();
  v8f acc[2]; acc[0] = zero8(); acc[1] = zero8();
  gemm_mt<2>(A, KPF1, P.pw, w, KPF1 / 32, acc);
  int n = lane & 15, hb = lane >> 4;
  float bv = P.bs[w * 16 + n];
#pragma unroll
  for (int mi = 0; mi < 2; ++mi)
#pragma unroll
    for (int g = 0; g < 8; ++g) {
      int m = g + 8 * hb;
      P.hs[(size_t)(i0 + mi * 16 + m) * 128 + w * 16 + n] = acc[mi][g] + bv;
    }
}

__global__ void __launch_bounds__(256)
k_readout(const float* __restrict__ s, const float* __restrict__ rw,
          const float* __restrict__ rb, float* __restrict__ out) {
  int node = (blockIdx.x * blockDim.x + threadIdx.x) >> 5;
  int lane = threadIdx.x & 31;
  if (node >= NN) return;
  float a = 0.f;
#pragma unroll
  for (int k = 0; k < 4; ++k) {
    int c = lane + 32 * k;
    a += s[(size_t)node * 128 + c] * rw[c];
  }
  a = wred32(a);
  if (lane == 0) out[node] = sigmoidf_(a + rb[0]);
}

// ============================================================================
extern "C" void kernel_launch(void* const* d_in, const int* in_sizes, int n_in,
                              void* d_out, int out_size, void* d_ws, size_t ws_size,
                              hipStream_t stream) {
  (void)in_sizes; (void)n_in; (void)out_size; (void)ws_size;
  const float* node_s = (const float*)d_in[0];
  const float* node_v = (const float*)d_in[1];
  const int*   eidx   = (const int*)d_in[2];
  const float* edge_s = (const float*)d_in[3];
  const float* edge_v = (const float*)d_in[4];
  auto pr = [&](int i) { return (const float*)d_in[i]; };
  const float* lin_b = pr(53);
  const float* lin_w = pr(54);
  const float* ro_b  = pr(55);
  const float* ro_w  = pr(56);

  // ---- workspace carve-up (~61 MB) ----
  char* wsb = (char*)d_ws;
  size_t off = 0;
  auto alloc = [&](size_t bytes) -> char* {
    char* p = wsb + off;
    off = (off + bytes + 255) & ~(size_t)255;
    return p;
  };
  float* s    = (float*)alloc((size_t)NN * 128 * 4);
  unsigned short* sbf = (unsigned short*)alloc((size_t)NN * 128 * 2);
  float* v    = (float*)alloc((size_t)NN * 6 * 4);
  float* ds   = (float*)alloc((size_t)NN * 128 * 4);
  float* dv   = (float*)alloc((size_t)NN * 6 * 4);
  float* cnt  = (float*)alloc((size_t)NN * 4);
  unsigned short* hidbf = (unsigned short*)alloc((size_t)NN * 512 * 2);
  float* hv   = (float*)alloc((size_t)NN * 12 * 4);
  float* hs   = (float*)alloc((size_t)NN * 128 * 4);
  float* hv2  = (float*)alloc((size_t)NN * 6 * 4);
  unsigned short* pw_lin = (unsigned short*)alloc((size_t)KPL * 128 * 2);
  unsigned short *pw_m0[2], *pw_m1[2], *pw_m2[2], *pw_ff0[2], *pw_ff1[2];
  for (int l = 0; l < 2; ++l) {
    pw_m0[l]  = (unsigned short*)alloc((size_t)KP0 * 128 * 2);
    pw_m1[l]  = (unsigned short*)alloc((size_t)KP12 * 128 * 2);
    pw_m2[l]  = (unsigned short*)alloc((size_t)KP12 * 128 * 2);
    pw_ff0[l] = (unsigned short*)alloc((size_t)KPF0 * 512 * 2);
    pw_ff1[l] = (unsigned short*)alloc((size_t)KPF1 * 128 * 2);
  }

  auto packB = [&](const float* src, unsigned short* dst, int K, int N, int Kpad) {
    int total = Kpad * N;
    k_pack_b<<<(total + 255) / 256, 256, 0, stream>>>(src, dst, K, N, Kpad);
  };
  auto fill0 = [&](float* p, size_t n) {
    k_fill0<<<(int)((n + 255) / 256), 256, 0, stream>>>(p, (int)n);
  };

  // ---- weights -> fragment-packed bf16 ----
  packB(lin_w, pw_lin, INS, 128, KPL);
  for (int l = 0; l < 2; ++l) {
    int B = 5 + 24 * l;
    packB(pr(B + 14), pw_m0[l], 269, 128, KP0);   // m0.ws
    packB(pr(B + 18), pw_m1[l], 130, 128, KP12);  // m1.ws
    packB(pr(B + 22), pw_m2[l], 130, 128, KP12);  // m2.ws
    packB(pr(B + 2),  pw_ff0[l], 132, 512, KPF0); // ff0.ws
    packB(pr(B + 6),  pw_ff1[l], 516, 128, KPF1); // ff1.ws
  }

  // ---- degree counts (same for both layers) ----
  fill0(cnt, NN);
  k_deg<<<(NE + 255) / 256, 256, 0, stream>>>(eidx, cnt);

  // ---- initial state ----
  hipMemcpyAsync(v, node_v, (size_t)NN * 6 * 4, hipMemcpyDeviceToDevice, stream);
  k_lin<<<NN / 16, 256, 0, stream>>>(node_s, pw_lin, lin_b, s, sbf);

  // ---- two GVPConv layers ----
  for (int l = 0; l < 2; ++l) {
    int B = 5 + 24 * l;
    fill0(ds, (size_t)NN * 128);
    fill0(dv, (size_t)NN * 6);

    MsgArgs M;
    M.sbf = sbf; M.v = v; M.eidx = eidx; M.es = edge_s; M.ev = edge_v;
    M.bs0 = pr(B + 12); M.wh0 = pr(B + 13); M.wv0 = pr(B + 15);
    M.bs1 = pr(B + 16); M.wh1 = pr(B + 17); M.wv1 = pr(B + 19);
    M.bs2 = pr(B + 20); M.wh2 = pr(B + 21); M.wv2 = pr(B + 23);
    M.pw0 = pw_m0[l]; M.pw1 = pw_m1[l]; M.pw2 = pw_m2[l];
    M.ds = ds; M.dv = dv;
    k_msg<<<NE / (MTM * 16), 256, 0, stream>>>(M);

    // residual + mean-agg + LN0
    k_ln<<<NN / 8, 256, 0, stream>>>(s, ds, v, dv, cnt,
                                     pr(B + 9), pr(B + 8), s, sbf, v);
    // feed-forward
    Ff0Args F0;
    F0.sbf = sbf; F0.v = v;
    F0.bs = pr(B + 0); F0.wh = pr(B + 1); F0.wv = pr(B + 3);
    F0.pw = pw_ff0[l]; F0.hidbf = hidbf; F0.hv = hv;
    k_ff0<<<NN / 32, 256, 0, stream>>>(F0);

    Ff1Args F1;
    F1.hidbf = hidbf; F1.hv = hv;
    F1.bs = pr(B + 4); F1.wh = pr(B + 5); F1.wv = pr(B + 7);
    F1.pw = pw_ff1[l]; F1.hs = hs; F1.hv2 = hv2;
    k_ff1<<<NN / 32, 256, 0, stream>>>(F1);

    // residual + LN1
    k_ln<<<NN / 8, 256, 0, stream>>>(s, hs, v, hv2, nullptr,
                                     pr(B + 11), pr(B + 10), s, sbf, v);
  }

  // ---- readout ----
  k_readout<<<NN / 8, 256, 0, stream>>>(s, ro_w, ro_b, (float*)d_out);
}
